// DeepSpeedMLP_50757923504330
// MI455X (gfx1250) — compile-verified, matrix operand order checked by reading, not run on hardware
//
#include <hip/hip_runtime.h>

// ---------------------------------------------------------------------------
// DeepSpeed MLP block on MI455X (gfx1250, wave32, WMMA).
// B=4, S=8192, H=1024, I=4096. ~550 GFLOP of GEMM, ~1.4GB HBM traffic.
// bf16 WMMA (v_wmma_f32_16x16x32_bf16) + global_load_async_to_lds_b128
// double-buffered pipeline; sigmoid-form GELU (v_exp_f32) epilogue.
// ---------------------------------------------------------------------------

typedef __attribute__((ext_vector_type(4)))  float  v4f;
typedef __attribute__((ext_vector_type(8)))  float  v8f;
typedef __attribute__((ext_vector_type(8)))  __bf16 v8bf;
typedef __attribute__((ext_vector_type(16))) __bf16 v16bf;
typedef __attribute__((ext_vector_type(4)))  __bf16 v4bf;

static constexpr int kB = 4, kS = 8192, kH = 1024, kI = 4096;
static constexpr int kM = kB * kS;  // 32768 rows
#define LN_EPS 1e-5f

// Direct global -> LDS async copy (16B per lane), GVS addressing:
//   lds[ldsOff] = mem[saddr + gOff]; tracked by ASYNCcnt.
__device__ __forceinline__ void async_copy_b128(unsigned ldsOff, unsigned gOff,
                                                unsigned long long base) {
  asm volatile("global_load_async_to_lds_b128 %0, %1, %2"
               :
               : "v"(ldsOff), "v"(gOff), "s"(base)
               : "memory");
}
__device__ __forceinline__ void wait_async_all() {
  asm volatile("s_wait_asynccnt 0" ::: "memory");
}

// ---------------------------------------------------------------------------
// Kernel 0: fp32 [K,N] -> bf16 [N,K] (transpose+convert). Weights are 16.8MB,
// fully L2-resident (192MB global L2), so naive indexing is fine.
// ---------------------------------------------------------------------------
__global__ __launch_bounds__(256) void transpose_to_bf16(
    const float* __restrict__ src, __bf16* __restrict__ dst, int K, int N) {
  long idx = (long)blockIdx.x * blockDim.x + threadIdx.x;  // over N*K
  int  n   = (int)(idx / K);
  int  k   = (int)(idx - (long)n * K);
  dst[idx] = (__bf16)src[(long)k * N + n];
}

// ---------------------------------------------------------------------------
// Kernel 1: residual_add = input + bias + residual (kept fp32 for epilogue),
// then LayerNorm -> x in bf16. One 256-thread block per row of H=1024.
// ---------------------------------------------------------------------------
__global__ __launch_bounds__(256) void fused_bias_res_ln(
    const float* __restrict__ input, const float* __restrict__ residual,
    const float* __restrict__ bias,  const float* __restrict__ gamma,
    const float* __restrict__ beta,  float* __restrict__ res_add,
    __bf16* __restrict__ xout) {
  const int  row  = blockIdx.x;
  const int  t    = threadIdx.x;
  const long base = (long)row * kH;
  const int  c    = t * 4;

  v4f in = *(const v4f*)(input + base + c);
  v4f rs = *(const v4f*)(residual + base + c);
  v4f bi = *(const v4f*)(bias + c);
  v4f r;
  float sum = 0.f, sq = 0.f;
#pragma unroll
  for (int j = 0; j < 4; ++j) {
    r[j] = in[j] + bi[j] + rs[j];
    sum += r[j];
    sq  += r[j] * r[j];
  }
  *(v4f*)(res_add + base + c) = r;

  __shared__ float s1[256], s2[256];
  s1[t] = sum; s2[t] = sq;
  __syncthreads();
#pragma unroll
  for (int off = 128; off > 0; off >>= 1) {
    if (t < off) { s1[t] += s1[t + off]; s2[t] += s2[t + off]; }
    __syncthreads();
  }
  const float mean = s1[0] * (1.f / kH);
  const float var  = s2[0] * (1.f / kH) - mean * mean;
  const float rstd = rsqrtf(var + LN_EPS);

  v4f g  = *(const v4f*)(gamma + c);
  v4f bb = *(const v4f*)(beta + c);
  v4bf o;
#pragma unroll
  for (int j = 0; j < 4; ++j) o[j] = (__bf16)((r[j] - mean) * rstd * g[j] + bb[j]);
  *(v4bf*)(xout + base + c) = o;
}

// ---------------------------------------------------------------------------
// WMMA GEMM: C[M,N] = A[M,K] * BT[N,K]^T (+epilogue).
// Block tile 128x256x32, 8 wave32 waves in a 2x4 grid, each wave owns a
// 64x64 output = 4x4 WMMA tiles (8 fragment loads feed 16 WMMAs per k-step).
// Operands flow global->LDS via global_load_async_to_lds_b128 (ASYNCcnt),
// double-buffered (60KB of the 320KB WGP LDS pool).
// Fragments follow the CDNA5 ISA 16-bit A/B layout: lane group (lane>>4)
// selects K-chunks {0..7,16..23} vs {8..15,24..31}, two 16B LDS reads each.
// EPI==0: +bias, sigmoid-form tanh-GELU, store bf16.
// EPI==1: +bias+resadd, store fp32.
// ---------------------------------------------------------------------------
template <int EPI>
__global__ __launch_bounds__(256) void wmma_gemm(
    const __bf16* __restrict__ A,   // [M,K] row-major bf16
    const __bf16* __restrict__ BT,  // [N,K] row-major bf16 (B transposed)
    const float* __restrict__ bias,     // [N]
    const float* __restrict__ resadd,   // [M,N] (EPI==1)
    __bf16* __restrict__ outB,          // [M,N] (EPI==0)
    float* __restrict__ outF,           // [M,N] (EPI==1)
    int M, int N, int K) {
  constexpr int BM = 128, BN = 256, BK = 32, LDSROW = BK + 8;
  __shared__ __align__(16) __bf16 As[2][BM][LDSROW];
  __shared__ __align__(16) __bf16 Bs[2][BN][LDSROW];

  const int tid  = threadIdx.x;
  const int lane = tid & 31;
  const int wave = tid >> 5;
  const int wm   = wave >> 2;              // 0..1 -> 64-row slab
  const int wn   = wave & 3;               // 0..3 -> 64-col slab
  const int lr   = lane & 15;
  const int kSel = (lane >> 4) << 3;       // 0 or 8

  const int blockN = blockIdx.x * BN;
  const int blockM = blockIdx.y * BM;

  // Cooperative async tile fill: A = 2 chunks/thread, B = 4 chunks/thread.
  const int row0 = tid >> 2;               // 0..63
  const int kc0  = (tid & 3) << 3;         // 0,8,16,24 (bf16 units)

  const unsigned long long Abase =
      (unsigned long long)(const void*)(A + (long)blockM * K);
  const unsigned long long Bbase =
      (unsigned long long)(const void*)(BT + (long)blockN * K);

  auto issueAsync = [&](int kt, int buf) {
    const long k0 = (long)kt * BK + kc0;
#pragma unroll
    for (int c = 0; c < 2; ++c) {
      const int r = row0 + c * 64;
      async_copy_b128((unsigned)(unsigned long)(uintptr_t)&As[buf][r][kc0],
                      (unsigned)(((long)r * K + k0) * 2), Abase);
    }
#pragma unroll
    for (int c = 0; c < 4; ++c) {
      const int r = row0 + c * 64;
      async_copy_b128((unsigned)(unsigned long)(uintptr_t)&Bs[buf][r][kc0],
                      (unsigned)(((long)r * K + k0) * 2), Bbase);
    }
  };

  v8f acc[4][4] = {};

  const int nk = K / BK;
  issueAsync(0, 0);
  wait_async_all();
  __syncthreads();

  for (int kt = 0; kt < nk; ++kt) {
    const int buf = kt & 1;
    if (kt + 1 < nk) {
      issueAsync(kt + 1, buf ^ 1);
      if (kt + 4 < nk) {  // deep L2 warm-up: global_prefetch_b8
        const long kp = (long)(kt + 4) * BK + kc0;
        __builtin_prefetch((const char*)(size_t)Abase + ((long)row0 * K + kp) * 2, 0, 1);
        __builtin_prefetch((const char*)(size_t)Bbase + ((long)row0 * K + kp) * 2, 0, 1);
      }
    }

    v16bf afrag[4], bfrag[4];
#pragma unroll
    for (int i = 0; i < 4; ++i) {
      const __bf16* p = &As[buf][wm * 64 + i * 16 + lr][0];
      v8bf lo = *(const v8bf*)(p + kSel);
      v8bf hi = *(const v8bf*)(p + kSel + 16);
      afrag[i] = __builtin_shufflevector(lo, hi, 0, 1, 2, 3, 4, 5, 6, 7, 8, 9,
                                         10, 11, 12, 13, 14, 15);
    }
#pragma unroll
    for (int j = 0; j < 4; ++j) {
      const __bf16* p = &Bs[buf][wn * 64 + j * 16 + lr][0];
      v8bf lo = *(const v8bf*)(p + kSel);
      v8bf hi = *(const v8bf*)(p + kSel + 16);
      bfrag[j] = __builtin_shufflevector(lo, hi, 0, 1, 2, 3, 4, 5, 6, 7, 8, 9,
                                         10, 11, 12, 13, 14, 15);
    }

#pragma unroll
    for (int i = 0; i < 4; ++i)
#pragma unroll
      for (int j = 0; j < 4; ++j)
        acc[i][j] = __builtin_amdgcn_wmma_f32_16x16x32_bf16(
            false, afrag[i], false, bfrag[j], (short)0, acc[i][j], false, false);

    __syncthreads();  // everyone done reading buf (DScnt waited by compiler)
    if (kt + 1 < nk) {
      wait_async_all();  // buf^1 fully landed in LDS
      __syncthreads();
    }
  }

  // Epilogue. C/D layout: VGPR r, lanes 0-15 -> M=r, lanes 16-31 -> M=8+r.
#pragma unroll
  for (int i = 0; i < 4; ++i) {
    const int rowBase = blockM + wm * 64 + i * 16 + ((lane >> 4) << 3);
#pragma unroll
    for (int j = 0; j < 4; ++j) {
      const int   col = blockN + wn * 64 + j * 16 + lr;
      const float bv  = bias[col];
#pragma unroll
      for (int r = 0; r < 8; ++r) {
        const long row = rowBase + r;
        const float v = acc[i][j][r] + bv;
        if (EPI == 0) {
          // tanh-GELU via sigmoid: 0.5*(1+tanh(u)) = 1/(1+exp(-2u)).
          const float u2 = 1.5957691216057308f * (v + 0.044715f * v * v * v);
          const float g  = v * __builtin_amdgcn_rcpf(1.0f + __expf(-u2));
          outB[row * N + col] = (__bf16)g;
        } else {
          outF[row * N + col] = v + resadd[row * N + col];
        }
      }
    }
  }
}

// ---------------------------------------------------------------------------
// Launch
// ---------------------------------------------------------------------------
extern "C" void kernel_launch(void* const* d_in, const int* in_sizes, int n_in,
                              void* d_out, int out_size, void* d_ws,
                              size_t ws_size, hipStream_t stream) {
  const float* input    = (const float*)d_in[0];
  const float* residual = (const float*)d_in[1];
  // d_in[2] residual_norm: unused by the reference math
  const float* bias     = (const float*)d_in[3];
  const float* attn_nw  = (const float*)d_in[4];
  const float* attn_nb  = (const float*)d_in[5];
  const float* inter_w  = (const float*)d_in[6];  // [H, I]
  const float* inter_b  = (const float*)d_in[7];
  const float* output_w = (const float*)d_in[8];  // [I, H]
  const float* output_b = (const float*)d_in[9];
  float*       out      = (float*)d_out;

  // Workspace layout (bytes):
  char*   ws      = (char*)d_ws;
  size_t  off     = 0;
  float*  res_add = (float*)(ws + off);  off += (size_t)kM * kH * 4;   // 134MB
  __bf16* xbf     = (__bf16*)(ws + off); off += (size_t)kM * kH * 2;   //  67MB
  __bf16* hbf     = (__bf16*)(ws + off); off += (size_t)kM * kI * 2;   // 268MB
  __bf16* w1T     = (__bf16*)(ws + off); off += (size_t)kI * kH * 2;   // 8.4MB
  __bf16* w2T     = (__bf16*)(ws + off); off += (size_t)kH * kI * 2;   // 8.4MB

  // 0) bf16 transposed weights: inter_w [H,I] -> w1T [I,H]; output_w [I,H] -> w2T [H,I]
  transpose_to_bf16<<<(kH * kI) / 256, 256, 0, stream>>>(inter_w, w1T, kH, kI);
  transpose_to_bf16<<<(kI * kH) / 256, 256, 0, stream>>>(output_w, w2T, kI, kH);

  // 1) fused bias+residual add + LayerNorm
  fused_bias_res_ln<<<kM, 256, 0, stream>>>(input, residual, bias, attn_nw,
                                            attn_nb, res_add, xbf);

  // 2) GEMM1 + bias + GELU -> h (bf16):  [32768,1024] x [1024,4096]
  wmma_gemm<0><<<dim3(kI / 256, kM / 128), 256, 0, stream>>>(
      xbf, w1T, inter_b, nullptr, hbf, nullptr, kM, kI, kH);

  // 3) GEMM2 + bias + residual_add -> out (fp32): [32768,4096] x [4096,1024]
  wmma_gemm<1><<<dim3(kH / 256, kM / 128), 256, 0, stream>>>(
      hbf, w2T, output_b, res_add, nullptr, out, kM, kH, kI);
}
